// FedMambaHC_20624432955861
// MI455X (gfx1250) — matrix operand order repaired and checked by reading, not compiled, not run on hardware
//
#include <hip/hip_runtime.h>
#include <hip/hip_bf16.h>
#include <math.h>

#define Bq    128
#define LEADS 12
#define Tn    1000
#define DM    128
#define DI    256
#define DSZ   16
#define DR    8
#define DCV   4
#define NBL   4
#define Lq    199
#define OUTF  256
#define NTOK  (Bq * Lq)   // 25472 = 199 * 128 = 398 * 64

typedef __attribute__((ext_vector_type(16))) _Float16 v16h;
typedef __attribute__((ext_vector_type(8)))  _Float16 v8h;
typedef __attribute__((ext_vector_type(8)))  float    v8f;

__device__ __forceinline__ float gelu_f(float x)  { return 0.5f * x * (1.0f + erff(x * 0.70710678118654752f)); }
__device__ __forceinline__ float silu_f(float x)  { return x / (1.0f + __expf(-x)); }
__device__ __forceinline__ float softplus_f(float x) { return (x > 20.0f) ? x : log1pf(__expf(x)); }

enum { EP_NONE = 0, EP_BIAS_SOFTPLUS = 1, EP_BIAS_GELU = 2, EP_ACC = 3, EP_ACC_BIAS = 4 };

#define LDS_STR 40   // halves; 80B row stride -> conflict-free b128 frag loads

__device__ __forceinline__ v8h cvt_pack8(float4 f0, float4 f1)
{
  union { v8h v; _Float16 h[8]; } pk;
  pk.h[0] = (_Float16)f0.x; pk.h[1] = (_Float16)f0.y;
  pk.h[2] = (_Float16)f0.z; pk.h[3] = (_Float16)f0.w;
  pk.h[4] = (_Float16)f1.x; pk.h[5] = (_Float16)f1.y;
  pk.h[6] = (_Float16)f1.z; pk.h[7] = (_Float16)f1.w;
  return pk.v;
}

// ---------------------------------------------------------------------------
// Fast GEMM: C[M,N] = op(A[M,K] @ W[N,K]^T + bias). Requires K%32==0,
// N == gridDim.x * NSUB*32, M == gridDim.y * MSUB*64.
// 256 threads = 8 waves (wy 0..3 over M, wx 0..1 over N); each wave computes
// (MSUB*16) x (NSUB*16) via MSUB x NSUB WMMAs per 32-K chunk.
// Software-pipelined with double-buffered LDS: global loads for chunk k+1 are
// issued before the WMMAs of chunk k; one barrier per K-iteration.
// ---------------------------------------------------------------------------
template<int MSUB, int NSUB>
__global__ __launch_bounds__(256) void gemm_wmma_fast(
    const float* __restrict__ A, int lda,
    const float* __restrict__ W,
    float* __restrict__ C, int ldc,
    const float* __restrict__ bias,
    int K, int mode)
{
  constexpr int MTILE = MSUB * 64;
  constexpr int NTILE = NSUB * 32;
  constexpr int AIT = MTILE / 64;     // A staging iterations (64 rows each)
  constexpr int WIT = NTILE / 64;     // W staging iterations
  __shared__ __align__(16) _Float16 As[2][MTILE * LDS_STR];
  __shared__ __align__(16) _Float16 Ws[2][NTILE * LDS_STR];

  const int tid  = threadIdx.x;
  const int lane = tid & 31;
  const int wave = tid >> 5;
  const int wy = wave >> 1;           // 0..3 : M group
  const int wx = wave & 1;            // 0..1 : N half
  const int m0 = blockIdx.y * MTILE;
  const int n0 = blockIdx.x * NTILE;

  v8f acc[MSUB][NSUB];
  #pragma unroll
  for (int ms = 0; ms < MSUB; ++ms)
    #pragma unroll
    for (int ns = 0; ns < NSUB; ++ns) acc[ms][ns] = (v8f){};

  const int lrow  = tid >> 2;         // 0..63
  const int lcol8 = (tid & 3) * 8;    // 0,8,16,24

  const float* aptr[AIT];
  const float* wptr[WIT];
  #pragma unroll
  for (int it = 0; it < AIT; ++it) aptr[it] = A + (size_t)(m0 + lrow + it * 64) * lda + lcol8;
  #pragma unroll
  for (int it = 0; it < WIT; ++it) wptr[it] = W + (size_t)(n0 + lrow + it * 64) * K + lcol8;

  float4 ar[AIT][2], wr[WIT][2];

  auto load_chunk = [&](int kk) {
    #pragma unroll
    for (int it = 0; it < AIT; ++it) {
      ar[it][0] = *(const float4*)(aptr[it] + kk);
      ar[it][1] = *(const float4*)(aptr[it] + kk + 4);
    }
    #pragma unroll
    for (int it = 0; it < WIT; ++it) {
      wr[it][0] = *(const float4*)(wptr[it] + kk);
      wr[it][1] = *(const float4*)(wptr[it] + kk + 4);
    }
  };
  auto store_tile = [&](int buf) {
    #pragma unroll
    for (int it = 0; it < AIT; ++it)
      *(v8h*)(&As[buf][(lrow + it * 64) * LDS_STR + lcol8]) = cvt_pack8(ar[it][0], ar[it][1]);
    #pragma unroll
    for (int it = 0; it < WIT; ++it)
      *(v8h*)(&Ws[buf][(lrow + it * 64) * LDS_STR + lcol8]) = cvt_pack8(wr[it][0], wr[it][1]);
  };
  auto compute = [&](int buf) {
    const int r  = lane & 15;
    const int kh = lane >> 4;
    v16h af[MSUB], bf[NSUB];
    #pragma unroll
    for (int ms = 0; ms < MSUB; ++ms) {
      // A frag: lane -> M row; elems 0..7 = K kh*8+0..7, elems 8..15 = K 16+kh*8+0..7
      const _Float16* ap = &As[buf][((wy * MSUB + ms) * 16 + r) * LDS_STR + kh * 8];
      af[ms] = __builtin_shufflevector(*(const v8h*)(ap), *(const v8h*)(ap + 16),
                                       0,1,2,3,4,5,6,7,8,9,10,11,12,13,14,15);
    }
    #pragma unroll
    for (int ns = 0; ns < NSUB; ++ns) {
      // B frag: lane -> N col; elems 0..7 = K kh*8+0..7, elems 8..15 = K 16+kh*8+0..7
      const _Float16* bp = &Ws[buf][((wx * NSUB + ns) * 16 + r) * LDS_STR + kh * 8];
      bf[ns] = __builtin_shufflevector(*(const v8h*)(bp), *(const v8h*)(bp + 16),
                                       0,1,2,3,4,5,6,7,8,9,10,11,12,13,14,15);
    }
    #pragma unroll
    for (int ms = 0; ms < MSUB; ++ms)
      #pragma unroll
      for (int ns = 0; ns < NSUB; ++ns)
        acc[ms][ns] = __builtin_amdgcn_wmma_f32_16x16x32_f16(false, af[ms], false, bf[ns],
                                                             (short)0, acc[ms][ns], false, false);
  };

  // Prologue: fetch + stage chunk 0
  load_chunk(0);
  store_tile(0);
  int cur = 0;
  for (int kk = 32; kk < K; kk += 32) {
    load_chunk(kk);          // async: consumed after this chunk's WMMAs
    __syncthreads();         // stores to As/Ws[cur] visible; reads of [1-cur] done
    compute(cur);
    store_tile(1 - cur);     // waits loadcnt here, after WMMA issue
    cur ^= 1;
  }
  __syncthreads();
  compute(cur);

  // Epilogue: C/D layout: VGPR rr -> M = (lane>>4)*8 + rr ; N = lane&15
  const int r  = lane & 15;
  const int mh = lane >> 4;
  #pragma unroll
  for (int ms = 0; ms < MSUB; ++ms) {
    #pragma unroll
    for (int ns = 0; ns < NSUB; ++ns) {
      const int n = n0 + (wx * NSUB + ns) * 16 + r;
      const float bv = bias ? bias[n] : 0.0f;
      #pragma unroll
      for (int rr = 0; rr < 8; ++rr) {
        const int m = m0 + (wy * MSUB + ms) * 16 + mh * 8 + rr;
        float v = acc[ms][ns][rr];
        float* dst = C + (size_t)m * ldc + n;
        if      (mode == EP_NONE)          *dst = v;
        else if (mode == EP_BIAS_SOFTPLUS) *dst = softplus_f(v + bv);
        else if (mode == EP_BIAS_GELU)     *dst = gelu_f(v + bv);
        else if (mode == EP_ACC)           *dst += v;
        else                               *dst += v + bv;   // EP_ACC_BIAS
      }
    }
  }
}

// ---------------------------------------------------------------------------
// Guarded GEMM (64 x NSUB*32 tile, single-buffer): handles ragged K / N via
// clamped addresses + select-zero (branch-free). Used for xproj (N=40) and
// dtproj (K=8) only.
// ---------------------------------------------------------------------------
template<int NSUB>
__global__ __launch_bounds__(256) void gemm_wmma_guard(
    const float* __restrict__ A, int lda,
    const float* __restrict__ W,
    float* __restrict__ C, int ldc,
    const float* __restrict__ bias,
    int Nfeat, int K, int mode)
{
  constexpr int NTILE = NSUB * 32;
  __shared__ __align__(16) _Float16 As[64 * LDS_STR];
  __shared__ __align__(16) _Float16 Ws[NTILE * LDS_STR];

  const int tid  = threadIdx.x;
  const int lane = tid & 31;
  const int wave = tid >> 5;
  const int wy = wave >> 1;
  const int wx = wave & 1;
  const int m0 = blockIdx.y * 64;
  const int n0 = blockIdx.x * NTILE;

  v8f acc[NSUB];
  #pragma unroll
  for (int ns = 0; ns < NSUB; ++ns) acc[ns] = (v8f){};

  const int lrow  = tid >> 2;
  const int lcol8 = (tid & 3) * 8;

  for (int kk = 0; kk < K; kk += 32) {
    {
      const float* rowp = A + (size_t)(m0 + lrow) * lda;
      union { v8h v; _Float16 h[8]; } pk;
      #pragma unroll
      for (int j = 0; j < 8; ++j) {
        int k  = kk + lcol8 + j;
        int kc = (k < K) ? k : (K - 1);       // clamped, always in-bounds
        float v = rowp[kc];
        pk.h[j] = (_Float16)((k < K) ? v : 0.0f);
      }
      *(v8h*)(As + lrow * LDS_STR + lcol8) = pk.v;
    }
    #pragma unroll
    for (int it = 0; it < NTILE / 64; ++it) {
      const int rowl = lrow + it * 64;
      const int n = n0 + rowl;
      const int nc = (n < Nfeat) ? n : (Nfeat - 1);
      const float* rowp = W + (size_t)nc * K;
      union { v8h v; _Float16 h[8]; } pk;
      #pragma unroll
      for (int j = 0; j < 8; ++j) {
        int k  = kk + lcol8 + j;
        int kc = (k < K) ? k : (K - 1);
        float v = rowp[kc];
        pk.h[j] = (_Float16)((n < Nfeat && k < K) ? v : 0.0f);
      }
      *(v8h*)(Ws + rowl * LDS_STR + lcol8) = pk.v;
    }
    __syncthreads();

    const int r  = lane & 15;
    const int kh = lane >> 4;
    const _Float16* ap = As + (wy * 16 + r) * LDS_STR + kh * 8;
    v16h af = __builtin_shufflevector(*(const v8h*)(ap), *(const v8h*)(ap + 16),
                                      0,1,2,3,4,5,6,7,8,9,10,11,12,13,14,15);
    v16h bf[NSUB];
    #pragma unroll
    for (int ns = 0; ns < NSUB; ++ns) {
      const _Float16* bp = Ws + ((wx * NSUB + ns) * 16 + r) * LDS_STR + kh * 8;
      bf[ns] = __builtin_shufflevector(*(const v8h*)(bp), *(const v8h*)(bp + 16),
                                       0,1,2,3,4,5,6,7,8,9,10,11,12,13,14,15);
    }
    #pragma unroll
    for (int ns = 0; ns < NSUB; ++ns)
      acc[ns] = __builtin_amdgcn_wmma_f32_16x16x32_f16(false, af, false, bf[ns],
                                                       (short)0, acc[ns], false, false);
    __syncthreads();
  }

  const int r  = lane & 15;
  const int mh = lane >> 4;
  #pragma unroll
  for (int ns = 0; ns < NSUB; ++ns) {
    const int n = n0 + (wx * NSUB + ns) * 16 + r;
    if (n >= Nfeat) continue;
    const float bv = bias ? bias[n] : 0.0f;
    #pragma unroll
    for (int rr = 0; rr < 8; ++rr) {
      const int m = m0 + wy * 16 + mh * 8 + rr;
      float v = acc[ns][rr];
      float* dst = C + (size_t)m * ldc + n;
      if      (mode == EP_NONE)          *dst = v;
      else if (mode == EP_BIAS_SOFTPLUS) *dst = softplus_f(v + bv);
      else if (mode == EP_BIAS_GELU)     *dst = gelu_f(v + bv);
      else if (mode == EP_ACC)           *dst += v;
      else                               *dst += v + bv;
    }
  }
}

// LayerNorm over last dim (128): wave-per-row, shuffle reduction (wave32), 8 rows/block.
__global__ __launch_bounds__(256) void ln_kernel(const float* __restrict__ X, float* __restrict__ Y,
                                                 const float* __restrict__ g, const float* __restrict__ b)
{
  const int lane = threadIdx.x & 31;
  const int wave = threadIdx.x >> 5;
  const size_t row = (size_t)blockIdx.x * 8 + wave;
  const float* xr = X + row * DM;
  float x0 = xr[lane], x1 = xr[lane + 32], x2 = xr[lane + 64], x3 = xr[lane + 96];
  float s = x0 + x1 + x2 + x3;
  #pragma unroll
  for (int m = 16; m >= 1; m >>= 1) s += __shfl_xor(s, m, 32);
  const float mean = s * (1.0f / DM);
  x0 -= mean; x1 -= mean; x2 -= mean; x3 -= mean;
  float q = x0 * x0 + x1 * x1 + x2 * x2 + x3 * x3;
  #pragma unroll
  for (int m = 16; m >= 1; m >>= 1) q += __shfl_xor(q, m, 32);
  const float rs = rsqrtf(q * (1.0f / DM) + 1e-5f);
  float* yr = Y + row * DM;
  yr[lane]      = x0 * rs * g[lane]      + b[lane];
  yr[lane + 32] = x1 * rs * g[lane + 32] + b[lane + 32];
  yr[lane + 64] = x2 * rs * g[lane + 64] + b[lane + 64];
  yr[lane + 96] = x3 * rs * g[lane + 96] + b[lane + 96];
}

// Front conv: stride 5, ksize 10, 12 -> 128 channels. out[b][d][l]
__global__ __launch_bounds__(256) void front_conv(const float* __restrict__ x, const float* __restrict__ pe_w,
                                                  const float* __restrict__ pe_b, float* __restrict__ out)
{
  const int b = blockIdx.x, d = blockIdx.y, l = threadIdx.x;
  if (l >= Lq) return;
  float acc = pe_b[d];
  const float* xb = x + (size_t)b * LEADS * Tn;
  const float* wd = pe_w + (size_t)d * LEADS * 10;
  #pragma unroll
  for (int c = 0; c < LEADS; ++c) {
    const float* xc = xb + c * Tn + l * 5;
    const float* wc = wd + c * 10;
    #pragma unroll
    for (int t = 0; t < 10; ++t) acc += xc[t] * wc[t];
  }
  out[((size_t)b * DM + d) * Lq + l] = acc;
}

// GroupNorm stats per (b, group): mean/rstd over 16 channels x 199
__global__ __launch_bounds__(256) void gn_stats(const float* __restrict__ conv, float* __restrict__ stats)
{
  __shared__ float ssum[256], ssq[256];
  const int b = blockIdx.x, g = blockIdx.y;
  float s = 0.0f, q = 0.0f;
  const float* base = conv + ((size_t)b * DM + g * 16) * Lq;
  for (int i = threadIdx.x; i < 16 * Lq; i += 256) { float v = base[i]; s += v; q += v * v; }
  ssum[threadIdx.x] = s; ssq[threadIdx.x] = q; __syncthreads();
  for (int t = 128; t > 0; t >>= 1) {
    if (threadIdx.x < t) { ssum[threadIdx.x] += ssum[threadIdx.x + t]; ssq[threadIdx.x] += ssq[threadIdx.x + t]; }
    __syncthreads();
  }
  if (threadIdx.x == 0) {
    const float inv = 1.0f / (16.0f * Lq);
    const float m = ssum[0] * inv;
    const float var = ssq[0] * inv - m * m;
    stats[(b * 8 + g) * 2 + 0] = m;
    stats[(b * 8 + g) * 2 + 1] = rsqrtf(var + 1e-5f);
  }
}

// Apply GN affine + GELU + transpose to [b][l][d] + pos
__global__ __launch_bounds__(128) void gn_apply(const float* __restrict__ conv, const float* __restrict__ stats,
    const float* __restrict__ gg, const float* __restrict__ gb, const float* __restrict__ pos,
    float* __restrict__ h)
{
  const int b = blockIdx.x, l = blockIdx.y, d = threadIdx.x;
  const int g = d >> 4;
  const float m = stats[(b * 8 + g) * 2], rs = stats[(b * 8 + g) * 2 + 1];
  float v = conv[((size_t)b * DM + d) * Lq + l];
  v = (v - m) * rs * gg[d] + gb[d];
  h[((size_t)b * Lq + l) * DM + d] = gelu_f(v) + pos[l * DM + d];
}

// Depthwise causal conv (DC=4) + bias + SiLU over xm = xz[:, :256]
__global__ __launch_bounds__(256) void dwconv_silu(const float* __restrict__ xz, const float* __restrict__ cw,
                                                   const float* __restrict__ cb, float* __restrict__ ussm)
{
  const int b = blockIdx.x, l = blockIdx.y, c = threadIdx.x;
  float acc = cb[c];
  const float* w = cw + c * DCV;
  #pragma unroll
  for (int t = 0; t < DCV; ++t) {
    const int ls = l - (DCV - 1) + t;
    if (ls >= 0) acc += xz[((size_t)b * Lq + ls) * (2 * DI) + c] * w[t];
  }
  ussm[((size_t)b * Lq + l) * DI + c] = silu_f(acc);
}

// Selective scan: one block per batch, one thread per channel d, 16 states in registers.
__global__ __launch_bounds__(256) void ssm_scan(const float* __restrict__ dbl, const float* __restrict__ delta,
    const float* __restrict__ ussm, const float* __restrict__ A_log, const float* __restrict__ Dp,
    float* __restrict__ y)
{
  __shared__ float Bs[DSZ], Cs[DSZ];
  const int b = blockIdx.x;
  const int d = threadIdx.x;
  float Aneg[DSZ], st[DSZ];
  #pragma unroll
  for (int s = 0; s < DSZ; ++s) { Aneg[s] = -__expf(A_log[d * DSZ + s]); st[s] = 0.0f; }
  const float Dv = Dp[d];
  for (int l = 0; l < Lq; ++l) {
    const size_t n = (size_t)b * Lq + l;
    __syncthreads();
    if (d < DSZ)            Bs[d]       = dbl[n * 40 + DR + d];
    else if (d < 2 * DSZ)   Cs[d - DSZ] = dbl[n * 40 + DR + DSZ + (d - DSZ)];
    __syncthreads();
    const float dt = delta[n * DI + d];
    const float u  = ussm[n * DI + d];
    const float du = dt * u;
    float acc = 0.0f;
    #pragma unroll
    for (int s = 0; s < DSZ; ++s) {
      st[s] = __expf(dt * Aneg[s]) * st[s] + du * Bs[s];
      acc += st[s] * Cs[s];
    }
    y[n * DI + d] = acc + u * Dv;
  }
}

// y *= silu(z), z = xz[:, 256:]
__global__ __launch_bounds__(256) void gate_kernel(float* __restrict__ y, const float* __restrict__ xz)
{
  const size_t i = (size_t)blockIdx.x * 256 + threadIdx.x;
  const size_t n = i / DI;
  const int    c = (int)(i % DI);
  y[i] *= silu_f(xz[n * (2 * DI) + DI + c]);
}

// Mean over L per (b, d)
__global__ __launch_bounds__(128) void mean_kernel(const float* __restrict__ u, float* __restrict__ out)
{
  const int b = blockIdx.x, d = threadIdx.x;
  float s = 0.0f;
  for (int l = 0; l < Lq; ++l) s += u[((size_t)b * Lq + l) * DM + d];
  out[b * DM + d] = s * (1.0f / Lq);
}

extern "C" void kernel_launch(void* const* d_in, const int* in_sizes, int n_in,
                              void* d_out, int out_size, void* d_ws, size_t ws_size,
                              hipStream_t stream)
{
  (void)in_sizes; (void)n_in; (void)out_size; (void)ws_size;
  const float* x       = (const float*)d_in[0];
  const float* pe_w    = (const float*)d_in[1];
  const float* pe_b    = (const float*)d_in[2];
  const float* gn_g    = (const float*)d_in[3];
  const float* gn_b    = (const float*)d_in[4];
  const float* pos     = (const float*)d_in[5];
  const float* ln1_g   = (const float*)d_in[6];
  const float* ln1_b   = (const float*)d_in[7];
  const float* in_w    = (const float*)d_in[8];
  const float* conv_w  = (const float*)d_in[9];
  const float* conv_b  = (const float*)d_in[10];
  const float* xproj_w = (const float*)d_in[11];
  const float* dtproj_w= (const float*)d_in[12];
  const float* dtproj_b= (const float*)d_in[13];
  const float* A_log   = (const float*)d_in[14];
  const float* Dp      = (const float*)d_in[15];
  const float* outw    = (const float*)d_in[16];
  const float* ln2_g   = (const float*)d_in[17];
  const float* ln2_b   = (const float*)d_in[18];
  const float* ffn_w1  = (const float*)d_in[19];
  const float* ffn_b1  = (const float*)d_in[20];
  const float* ffn_w2  = (const float*)d_in[21];
  const float* ffn_b2  = (const float*)d_in[22];
  const float* lnf_g   = (const float*)d_in[23];
  const float* lnf_b   = (const float*)d_in[24];
  const float* head_w  = (const float*)d_in[25];
  const float* head_b  = (const float*)d_in[26];

  char* ws = (char*)d_ws;
  size_t off = 0;
  auto alloc = [&](size_t nfloats) { float* p = (float*)(ws + off); off += nfloats * sizeof(float); return p; };
  float* h      = alloc((size_t)NTOK * DM);       // residual stream [B*L, 128]
  float* bufU   = alloc((size_t)NTOK * DM);       // conv_tmp / LN output
  float* bufXZ  = alloc((size_t)NTOK * 2 * DI);   // xz / ffn hidden
  float* bufUS  = alloc((size_t)NTOK * DI);       // u_ssm
  float* bufDL  = alloc((size_t)NTOK * DI);       // delta
  float* bufY   = alloc((size_t)NTOK * DI);       // scan output
  float* bufDB  = alloc((size_t)NTOK * 40);       // dt|B|C projections
  float* stats  = alloc((size_t)Bq * 8 * 2);
  float* pooled = alloc((size_t)Bq * DM);

  // Front end
  front_conv<<<dim3(Bq, DM), 256, 0, stream>>>(x, pe_w, pe_b, bufU);
  gn_stats  <<<dim3(Bq, 8),  256, 0, stream>>>(bufU, stats);
  gn_apply  <<<dim3(Bq, Lq), 128, 0, stream>>>(bufU, stats, gn_g, gn_b, pos, h);

  const int MT128 = NTOK / 128;  // 199 (fast-path M tiles)
  const int MT64  = NTOK / 64;   // 398 (guarded M tiles)
  const int LNB   = NTOK / 8;    // ln blocks (8 rows each)
  for (int i = 0; i < NBL; ++i) {
    ln_kernel<<<LNB, 256, 0, stream>>>(h, bufU, ln1_g + i * DM, ln1_b + i * DM);
    // xz = u @ in_w^T  [N, 512]
    gemm_wmma_fast<2, 4><<<dim3(512 / 128, MT128), 256, 0, stream>>>(bufU, DM, in_w + (size_t)i * 2 * DI * DM,
                                                                     bufXZ, 2 * DI, nullptr, DM, EP_NONE);
    dwconv_silu<<<dim3(Bq, Lq), 256, 0, stream>>>(bufXZ, conv_w + (size_t)i * DI * DCV, conv_b + (size_t)i * DI, bufUS);
    // dbl = u_ssm @ xproj^T  [N, 40]  (guarded: N=40)
    gemm_wmma_guard<2><<<dim3(1, MT64), 256, 0, stream>>>(bufUS, DI, xproj_w + (size_t)i * 40 * DI,
                                                          bufDB, 40, nullptr, 40, DI, EP_NONE);
    // delta = softplus(dt @ dtproj^T + b)  [N, 256]  (guarded: K=8; A = dbl, lda=40)
    gemm_wmma_guard<2><<<dim3(DI / 64, MT64), 256, 0, stream>>>(bufDB, 40, dtproj_w + (size_t)i * DI * DR,
                                                                bufDL, DI, dtproj_b + (size_t)i * DI, DI, DR, EP_BIAS_SOFTPLUS);
    ssm_scan<<<Bq, DI, 0, stream>>>(bufDB, bufDL, bufUS, A_log + (size_t)i * DI * DSZ, Dp + (size_t)i * DI, bufY);
    gate_kernel<<<((size_t)NTOK * DI) / 256, 256, 0, stream>>>(bufY, bufXZ);
    // h += y @ outw^T  [N, 128]
    gemm_wmma_fast<2, 4><<<dim3(DM / 128, MT128), 256, 0, stream>>>(bufY, DI, outw + (size_t)i * DM * DI,
                                                                    h, DM, nullptr, DI, EP_ACC);
    ln_kernel<<<LNB, 256, 0, stream>>>(h, bufU, ln2_g + i * DM, ln2_b + i * DM);
    // hidden = gelu(u @ w1^T + b1)  [N, 512]
    gemm_wmma_fast<2, 4><<<dim3(512 / 128, MT128), 256, 0, stream>>>(bufU, DM, ffn_w1 + (size_t)i * 4 * DM * DM,
                                                                     bufXZ, 4 * DM, ffn_b1 + (size_t)i * 4 * DM, DM, EP_BIAS_GELU);
    // h += hidden @ w2^T + b2  [N, 128]
    gemm_wmma_fast<2, 4><<<dim3(DM / 128, MT128), 256, 0, stream>>>(bufXZ, 4 * DM, ffn_w2 + (size_t)i * DM * 4 * DM,
                                                                    h, DM, ffn_b2 + (size_t)i * DM, 4 * DM, EP_ACC_BIAS);
  }

  // Final LN + mean pool + head
  ln_kernel<<<LNB, 256, 0, stream>>>(h, bufU, lnf_g, lnf_b);
  mean_kernel<<<Bq, 128, 0, stream>>>(bufU, pooled);
  gemm_wmma_fast<2, 4><<<dim3(OUTF / 128, Bq / 128), 256, 0, stream>>>(pooled, DM, head_w,
                                                                       (float*)d_out, OUTF, head_b, DM, EP_BIAS_GELU);
}
